// Q_Fun_37228776522458
// MI455X (gfx1250) — compile-verified, matrix operand order checked by reading, not run on hardware
//
#include <hip/hip_runtime.h>
#include <hip/hip_bf16.h>

typedef __attribute__((ext_vector_type(8)))  float  v8f;
typedef __attribute__((ext_vector_type(16))) __bf16 v16bf;

// fragment register buffer: 8 dwords == 16 bf16 == one WMMA A/B operand
typedef union {
    unsigned int u32[8];
    __bf16       b[16];
    v16bf        v;
} frag_t;

__device__ __forceinline__ unsigned int pack_bf16x2(float a, float b) {
    union { __bf16 h[2]; unsigned int u; } p;
    p.h[0] = (__bf16)a;            // native v_cvt RNE
    p.h[1] = (__bf16)b;
    return p.u;
}

// ---------------------------------------------------------------------------
// WMMA GEMM: C[nrows,128] (beta? += : =) A[nrows,128] @ W[128,128]
// A,W,C f32 row-major; A/W -> bf16 (native cvt), f32 accumulate.
// Block = 256 thr = 8 waves; each wave owns a 16x128 strip; block tile 128x128.
// B fragments are double-buffered to overlap ds_load latency with the
// matrix pipe instead of s_wait_dscnt 0 before every WMMA.
// ---------------------------------------------------------------------------
__global__ __launch_bounds__(256) void gemm128_bf16_wmma(
    const float* __restrict__ A, const float* __restrict__ W,
    float* __restrict__ C, int nrows, int beta, int reluA)
{
    // transposed weights, bf16, stored as packed dwords: Wt32[n*64 + k/2]
    __shared__ unsigned int Wt32[128 * 64];    // 32 KB
    const int tid = threadIdx.x;

    for (int idx = tid; idx < 128 * 64; idx += 256) {
        int n  = idx >> 6;
        int k0 = (idx & 63) * 2;
        Wt32[idx] = pack_bf16x2(W[k0 * 128 + n], W[(k0 + 1) * 128 + n]);
    }
    __syncthreads();

    const int wave = tid >> 5;
    const int lane = tid & 31;
    const int m16  = lane & 15;
    const int half = lane >> 4;
    const int rowBase = blockIdx.x * 128 + wave * 16;

    v8f c[8];
    if (beta) {
        #pragma unroll
        for (int nt = 0; nt < 8; ++nt) {
            int col = nt * 16 + m16;
            #pragma unroll
            for (int r = 0; r < 8; ++r) {
                int row = rowBase + r + 8 * half;
                row = row < nrows ? row : nrows - 1;
                c[nt][r] = C[(size_t)row * 128 + col];
            }
        }
    } else {
        #pragma unroll
        for (int nt = 0; nt < 8; ++nt)
            #pragma unroll
            for (int r = 0; r < 8; ++r) c[nt][r] = 0.0f;
    }

    int arow = rowBase + m16;
    if (arow >= nrows) arow = nrows - 1;
    const float* Arow = A + (size_t)arow * 128;

    #pragma unroll
    for (int kt = 0; kt < 4; ++kt) {
        // ---- A fragment: row m16, K packed 2/dword per ISA layout ----
        frag_t afrag;
        #pragma unroll
        for (int v = 0; v < 8; ++v) {
            int kk = kt * 32 + ((v & 4) ? 16 : 0) + half * 8 + (v & 3) * 2;
            float a0 = Arow[kk], a1 = Arow[kk + 1];
            if (reluA) { a0 = fmaxf(a0, 0.0f); a1 = fmaxf(a1, 0.0f); }
            afrag.u32[v] = pack_bf16x2(a0, a1);
        }

        // ---- B fragments from LDS, double-buffered across nt ----
        const int bbase = m16 * 64 + half * 4 + kt * 16;   // dword index
        frag_t bf[2];
        #pragma unroll
        for (int v = 0; v < 8; ++v)
            bf[0].u32[v] = Wt32[bbase + ((v & 4) ? 8 : 0) + (v & 3)];

        #pragma unroll
        for (int nt = 0; nt < 8; ++nt) {
            if (nt < 7) {
                const int nb = bbase + (nt + 1) * 1024;
                #pragma unroll
                for (int v = 0; v < 8; ++v)
                    bf[(nt + 1) & 1].u32[v] = Wt32[nb + ((v & 4) ? 8 : 0) + (v & 3)];
            }
            c[nt] = __builtin_amdgcn_wmma_f32_16x16x32_bf16(
                        false, afrag.v, false, bf[nt & 1].v,
                        (short)0, c[nt], false, false);
        }
    }

    #pragma unroll
    for (int nt = 0; nt < 8; ++nt) {
        int col = nt * 16 + m16;
        #pragma unroll
        for (int r = 0; r < 8; ++r) {
            int row = rowBase + r + 8 * half;
            if (row < nrows) C[(size_t)row * 128 + col] = c[nt][r];
        }
    }
}

// ---------------------------------------------------------------------------
// Elementwise / graph kernels
// ---------------------------------------------------------------------------
__global__ void init_kernel(float* dinv, float* pool0, int n) {
    int i = blockIdx.x * 256 + threadIdx.x;
    if (i < n)   dinv[i]  = 1.0f;       // self-loop contributes 1 to degree
    if (i < 128) pool0[i] = 0.0f;
}

__global__ void deg_kernel(const int* __restrict__ col, float* deg, int E) {
    int e = blockIdx.x * 256 + threadIdx.x;
    if (e < E) atomicAdd(&deg[col[e]], 1.0f);
}

__global__ void rsqrt_kernel(float* dinv, int n) {
    int i = blockIdx.x * 256 + threadIdx.x;
    if (i < n) dinv[i] = rsqrtf(dinv[i]);   // deg >= 1 always (self loop)
}

__global__ void lin1_kernel(const float* __restrict__ x, const float* __restrict__ w1,
                            const float* __restrict__ b1, float* __restrict__ x1, int n) {
    int idx = blockIdx.x * 256 + threadIdx.x;
    if (idx >= n * 128) return;
    int i = idx >> 7, j = idx & 127;
    x1[idx] = b1[j] + x[i * 3 + 0] * w1[j]
                    + x[i * 3 + 1] * w1[128 + j]
                    + x[i * 3 + 2] * w1[256 + j];
}

__global__ void self_init_kernel(const float* __restrict__ hW,
                                 const float* __restrict__ dinv,
                                 float* __restrict__ acc, int n) {
    int idx = blockIdx.x * 256 + threadIdx.x;
    if (idx >= n * 128) return;
    float d = dinv[idx >> 7];
    acc[idx] = hW[idx] * d * d;
}

// one wave (32 lanes) per edge; each lane: float4 gather + 4 atomic f32 adds (L2)
__global__ __launch_bounds__(256) void edge_scatter_kernel(
    const float* __restrict__ hW, const float* __restrict__ dinv,
    const int* __restrict__ row, const int* __restrict__ col,
    float* __restrict__ acc, int E)
{
    int e = blockIdx.x * 8 + (threadIdx.x >> 5);
    if (e >= E) return;
    int lane = threadIdx.x & 31;
    int r = row[e], c = col[e];
    float norm = dinv[r] * dinv[c];
    const float4 v = ((const float4*)(hW + (size_t)r * 128))[lane];
    float* dst = acc + (size_t)c * 128 + lane * 4;
    atomicAdd(dst + 0, v.x * norm);
    atomicAdd(dst + 1, v.y * norm);
    atomicAdd(dst + 2, v.z * norm);
    atomicAdd(dst + 3, v.w * norm);
}

__global__ void bias_act_kernel(float* __restrict__ buf, const float* __restrict__ b,
                                int n, int do_relu) {
    int idx = blockIdx.x * 256 + threadIdx.x;
    if (idx >= n * 128) return;
    float v = buf[idx] + b[idx & 127];
    buf[idx] = do_relu ? fmaxf(v, 0.0f) : v;
}

__global__ __launch_bounds__(256) void pool_kernel(
    const float* __restrict__ nv, const int* __restrict__ sel,
    float* __restrict__ pool0, int n)
{
    __shared__ float loc[128];
    if (threadIdx.x < 128) loc[threadIdx.x] = 0.0f;
    __syncthreads();
    int total = n * 128;
    for (int idx = blockIdx.x * 256 + threadIdx.x; idx < total; idx += gridDim.x * 256) {
        if (sel[idx >> 7] == 0) atomicAdd(&loc[idx & 127], nv[idx]);
    }
    __syncthreads();
    if (threadIdx.x < 128) atomicAdd(&pool0[threadIdx.x], loc[threadIdx.x]);
}

// single block, 128 threads: g = pool0@w6+b6 ; gterm = b5 + relu(g)@w5_top
__global__ __launch_bounds__(128) void gg_kernel(
    const float* __restrict__ pool0, const float* __restrict__ w6,
    const float* __restrict__ b6, const float* __restrict__ w5,
    const float* __restrict__ b5, float* __restrict__ gterm)
{
    __shared__ float gr[128];
    int j = threadIdx.x;
    float acc = b6[j];
    for (int k = 0; k < 128; ++k) acc += pool0[k] * w6[k * 128 + j];
    gr[j] = fmaxf(acc, 0.0f);
    __syncthreads();
    float t = b5[j];
    for (int k = 0; k < 128; ++k) t += gr[k] * w5[k * 128 + j];
    gterm[j] = t;
}

// one wave per node: q[i] = b8 + sum_j relu(C2[i,j]+gterm[j]) * w8[j]
__global__ __launch_bounds__(256) void q_kernel(
    const float* __restrict__ C2, const float* __restrict__ gterm,
    const float* __restrict__ w8, const float* __restrict__ b8,
    float* __restrict__ q, int n)
{
    int node = blockIdx.x * 8 + (threadIdx.x >> 5);
    if (node >= n) return;
    int lane = threadIdx.x & 31;
    const float4 v  = ((const float4*)(C2 + (size_t)node * 128))[lane];
    const float4 g4 = ((const float4*)gterm)[lane];
    const float4 w4 = ((const float4*)w8)[lane];
    float s = fmaxf(v.x + g4.x, 0.0f) * w4.x
            + fmaxf(v.y + g4.y, 0.0f) * w4.y
            + fmaxf(v.z + g4.z, 0.0f) * w4.z
            + fmaxf(v.w + g4.w, 0.0f) * w4.w;
    #pragma unroll
    for (int off = 16; off > 0; off >>= 1) s += __shfl_down(s, off, 32);
    if (lane == 0) q[node] = s + b8[0];
}

// ---------------------------------------------------------------------------
extern "C" void kernel_launch(void* const* d_in, const int* in_sizes, int n_in,
                              void* d_out, int out_size, void* d_ws, size_t ws_size,
                              hipStream_t stream) {
    const float* x    = (const float*)d_in[0];
    const int*   ei   = (const int*)  d_in[1];
    const int*   sel  = (const int*)  d_in[2];
    const float* w1   = (const float*)d_in[3];
    const float* b1   = (const float*)d_in[4];
    const float* wc1  = (const float*)d_in[5];
    const float* bc1  = (const float*)d_in[6];
    const float* wc2  = (const float*)d_in[7];
    const float* bc2  = (const float*)d_in[8];
    const float* w2   = (const float*)d_in[9];
    const float* b2   = (const float*)d_in[10];
    const float* w5   = (const float*)d_in[11];
    const float* b5   = (const float*)d_in[12];
    const float* w6   = (const float*)d_in[13];
    const float* b6   = (const float*)d_in[14];
    const float* w8   = (const float*)d_in[15];
    const float* b8   = (const float*)d_in[16];

    const int N = in_sizes[0] / 3;
    const int E = in_sizes[1] / 2;
    const int* erow = ei;
    const int* ecol = ei + E;

    float* qout = (float*)d_out;
    float* NV   = qout + N;               // nodes_vec written straight to output

    const size_t nh = (size_t)N * 128;
    float* ws    = (float*)d_ws;
    float* x1    = ws;
    float* hW    = ws + nh;
    float* x2    = ws + 2 * nh;
    float* x3    = ws + 3 * nh;
    float* dinv  = ws + 4 * nh;
    float* pool0 = dinv + N;
    float* gterm = pool0 + 128;

    const int Gel = (int)((nh + 255) / 256);      // elementwise over N*128
    const int Gn  = (N + 255) / 256;
    const int Ge  = (E + 255) / 256;
    const int Ges = (E + 7) / 8;                  // wave-per-edge
    const int Gg  = (N + 127) / 128;              // GEMM 128-row tiles
    const int Gq  = (N + 7) / 8;

    // degree / dinv
    init_kernel <<<Gn, 256, 0, stream>>>(dinv, pool0, N);
    deg_kernel  <<<Ge, 256, 0, stream>>>(ecol, dinv, E);
    rsqrt_kernel<<<Gn, 256, 0, stream>>>(dinv, N);

    // lin1
    lin1_kernel<<<Gel, 256, 0, stream>>>(x, w1, b1, x1, N);

    // conv1: hW = x1@wc1 ; acc = D^-1/2 (A+I) D^-1/2 hW ; x2 = relu(acc + bc1)
    gemm128_bf16_wmma <<<Gg, 256, 0, stream>>>(x1, wc1, hW, N, 0, 0);
    self_init_kernel  <<<Gel, 256, 0, stream>>>(hW, dinv, x2, N);
    edge_scatter_kernel<<<Ges, 256, 0, stream>>>(hW, dinv, erow, ecol, x2, E);
    bias_act_kernel   <<<Gel, 256, 0, stream>>>(x2, bc1, N, 1);

    // conv2
    gemm128_bf16_wmma <<<Gg, 256, 0, stream>>>(x2, wc2, hW, N, 0, 0);
    self_init_kernel  <<<Gel, 256, 0, stream>>>(hW, dinv, x3, N);
    edge_scatter_kernel<<<Ges, 256, 0, stream>>>(hW, dinv, erow, ecol, x3, E);
    bias_act_kernel   <<<Gel, 256, 0, stream>>>(x3, bc2, N, 1);

    // nodes_vec = [x1|x2|x3] @ w2 + b2  (three accumulated K=128 GEMMs)
    gemm128_bf16_wmma<<<Gg, 256, 0, stream>>>(x1, w2,              NV, N, 0, 0);
    gemm128_bf16_wmma<<<Gg, 256, 0, stream>>>(x2, w2 + 128 * 128,  NV, N, 1, 0);
    gemm128_bf16_wmma<<<Gg, 256, 0, stream>>>(x3, w2 + 256 * 128,  NV, N, 1, 0);
    bias_act_kernel  <<<Gel, 256, 0, stream>>>(NV, b2, N, 0);

    // pooled branch: pool0 -> g -> gterm = b5 + relu(g)@w5_top
    pool_kernel<<<512, 256, 0, stream>>>(NV, sel, pool0, N);
    gg_kernel  <<<1, 128, 0, stream>>>(pool0, w6, b6, w5, b5, gterm);

    // h-pre = relu(nodes_vec) @ w5_bot ; q = b8 + relu(h-pre + gterm) . w8
    gemm128_bf16_wmma<<<Gg, 256, 0, stream>>>(NV, w5 + 128 * 128, hW, N, 0, 1);
    q_kernel<<<Gq, 256, 0, stream>>>(hW, gterm, w8, b8, qout, N);
}
